// GATNet_20117626814682
// MI455X (gfx1250) — matrix operand order repaired and checked by reading, not compiled, not run on hardware
//
#include <hip/hip_runtime.h>
#include <hip/hip_bf16.h>
#include <hip/hip_fp16.h>

// ---------------------------------------------------------------------------
// GATNet on gfx1250 (MI455X): 3x GATConv.
// GEMMs: f16 WMMA (16x16x32) with pre-packed f16 operands (A row-major,
// B transposed) so fragments load as global_load_b128 pairs, no cvt in loop.
// Edge softmax: monotone-uint atomicMax + float atomicAdd (L2-resident).
// ---------------------------------------------------------------------------

typedef __attribute__((ext_vector_type(16))) _Float16 v16h;
typedef __attribute__((ext_vector_type(8)))  _Float16 v8h;
typedef __attribute__((ext_vector_type(8)))  float    v8f;

#define NN      50000
#define NE      800000
#define IN_CH   128
#define HID_CH  256
#define OUT_CH  16
#define GSZ     50
#define NGRAPH  (NN / GSZ)          // 1000

// monotone-uint encoding of float so unsigned atomicMax == float max
__device__ __forceinline__ unsigned mono_enc(float f) {
    unsigned u = __float_as_uint(f);
    return (u & 0x80000000u) ? ~u : (u | 0x80000000u);
}
__device__ __forceinline__ float mono_dec(unsigned e) {
    unsigned u = (e & 0x80000000u) ? (e ^ 0x80000000u) : ~e;
    return __uint_as_float(u);
}
// host-side constant: mono_enc(-inf) = ~0xFF800000 = 0x007FFFFF
#define MONO_NEG_INF 0x007FFFFFu

// ---------------------------------------------------------------------------
// Fragment loaders from pre-packed f16 (b128 pairs).
// A (16x32, X16 row-major): lane -> row (l&15); half -> K block {0..15} vs
// {16..31}.  B (32x16, Wt col-major = Wt[n*K+k]): lane -> col (l&15);
// half -> K block {0..15} vs {16..31}, contiguous in Wt.
// ---------------------------------------------------------------------------
__device__ __forceinline__ v16h load_a16(const _Float16* __restrict__ p)
{
    v8h lo = *reinterpret_cast<const v8h*>(p);        // K base .. +7
    v8h hi = *reinterpret_cast<const v8h*>(p + 8);    // K base+8 .. +15
    return __builtin_shufflevector(lo, hi, 0,1,2,3,4,5,6,7,8,9,10,11,12,13,14,15);
}
__device__ __forceinline__ v16h load_b16(const _Float16* __restrict__ p)
{
    v8h lo = *reinterpret_cast<const v8h*>(p);        // K base .. +7
    v8h hi = *reinterpret_cast<const v8h*>(p + 8);    // K base+8 .. +15
    return __builtin_shufflevector(lo, hi, 0,1,2,3,4,5,6,7,8,9,10,11,12,13,14,15);
}

// ---------------------------------------------------------------------------
// GEMM: H[N,F] = X16[N,K] @ W[K,F] (Wt = f16 transposed weights, Wt[n*K+k]).
// One wave per 16x(16*CT) output tile; K fully unrolled.
// ---------------------------------------------------------------------------
template<int K, int F, int CT>
__global__ __launch_bounds__(256) void gemm_wmma_f16(
    const _Float16* __restrict__ X, const _Float16* __restrict__ Wt,
    float* __restrict__ H, int N)
{
    constexpr int colGroups = F / (16 * CT);
    const int wave = (int)((blockIdx.x * blockDim.x + threadIdx.x) >> 5);
    const int lane = (int)(threadIdx.x & 31);
    const int tiles = (N >> 4) * colGroups;
    if (wave >= tiles) return;

    const int rowT = wave / colGroups;
    const int colG = wave - rowT * colGroups;
    const int row0 = rowT << 4;
    const int col0 = colG * (16 * CT);

    const int m    = lane & 15;
    const int half = lane >> 4;

    v8f c[CT];
    #pragma unroll
    for (int t = 0; t < CT; ++t) c[t] = (v8f){};

    const _Float16* xrow = X + (size_t)(row0 + m) * K + half * 16;
    const _Float16* wrow[CT];
    #pragma unroll
    for (int t = 0; t < CT; ++t)
        wrow[t] = Wt + (size_t)(col0 + 16 * t + m) * K + half * 16;

    #pragma unroll
    for (int k0 = 0; k0 < K; k0 += 32) {
        v16h a = load_a16(xrow + k0);
        #pragma unroll
        for (int t = 0; t < CT; ++t) {
            v16h b = load_b16(wrow[t] + k0);
            c[t] = __builtin_amdgcn_wmma_f32_16x16x32_f16(
                       false, a, false, b, (short)0, c[t], false, false);
        }
    }

    // C/D layout: VGPR v -> row (v + half*8), lane%16 -> column
    #pragma unroll
    for (int t = 0; t < CT; ++t)
        #pragma unroll
        for (int v = 0; v < 8; ++v)
            H[(size_t)(row0 + v + half * 8) * F + col0 + 16 * t + m] = c[t][v];
}

// ---------------------------------------------------------------------------
// Operand packing
// ---------------------------------------------------------------------------
__global__ void pack_w16t(const float* __restrict__ W, _Float16* __restrict__ Wt,
                          int K, int F)
{
    int i = (int)(blockIdx.x * blockDim.x + threadIdx.x);
    if (i >= K * F) return;
    int k = i / F, n = i - k * F;
    Wt[(size_t)n * K + k] = (_Float16)W[i];
}
__global__ void pack_f16(const float* __restrict__ X, _Float16* __restrict__ Y, int n)
{
    int i = (int)(blockIdx.x * blockDim.x + threadIdx.x);
    if (i < n) Y[i] = (_Float16)X[i];
}

// ---------------------------------------------------------------------------
// Per-node attention scores: s_src = H.a_s, s_dst = H.a_d (one wave per node)
// ---------------------------------------------------------------------------
__global__ __launch_bounds__(256) void node_scores(
    const float* __restrict__ H, const float* __restrict__ avs,
    const float* __restrict__ avd, float* __restrict__ ss,
    float* __restrict__ sd, int N, int F)
{
    const int node = (int)((blockIdx.x * blockDim.x + threadIdx.x) >> 5);
    const int lane = (int)(threadIdx.x & 31);
    if (node >= N) return;
    float accS = 0.f, accD = 0.f;
    for (int i = lane; i < F; i += 32) {
        float h = H[(size_t)node * F + i];
        accS += h * avs[i];
        accD += h * avd[i];
    }
    #pragma unroll
    for (int o = 16; o > 0; o >>= 1) {
        accS += __shfl_xor(accS, o, 32);
        accD += __shfl_xor(accD, o, 32);
    }
    if (lane == 0) { ss[node] = accS; sd[node] = accD; }
}

// ---------------------------------------------------------------------------
// Fills
// ---------------------------------------------------------------------------
__global__ void fill_f32(float* p, float v, int n) {
    int i = (int)(blockIdx.x * blockDim.x + threadIdx.x);
    if (i < n) p[i] = v;
}
__global__ void fill_u32(unsigned* p, unsigned v, int n) {
    int i = (int)(blockIdx.x * blockDim.x + threadIdx.x);
    if (i < n) p[i] = v;
}

// ---------------------------------------------------------------------------
// Edge pass 1: e = leaky_relu(ss[src]+sd[dst], 0.2); per-dst segment max
// ---------------------------------------------------------------------------
__global__ void edge_score_max(
    const int* __restrict__ src, const int* __restrict__ dst,
    const float* __restrict__ ss, const float* __restrict__ sd,
    float* __restrict__ ebuf, unsigned* __restrict__ mmax, int E)
{
    int i = (int)(blockIdx.x * blockDim.x + threadIdx.x);
    if (i >= E) return;
    float e = ss[src[i]] + sd[dst[i]];
    e = (e > 0.f) ? e : 0.2f * e;
    ebuf[i] = e;
    atomicMax(mmax + dst[i], mono_enc(e));
}

// ---------------------------------------------------------------------------
// Edge pass 2: ex = exp(e - m[dst]); denom[dst] += ex
// ---------------------------------------------------------------------------
__global__ void edge_exp_sum(
    const int* __restrict__ dst, const unsigned* __restrict__ mmax,
    float* __restrict__ ebuf, float* __restrict__ denom, int E)
{
    int i = (int)(blockIdx.x * blockDim.x + threadIdx.x);
    if (i >= E) return;
    int d = dst[i];
    float v = __expf(ebuf[i] - mono_dec(mmax[d]));
    ebuf[i] = v;
    atomicAdd(denom + d, v);
}

// ---------------------------------------------------------------------------
// Edge pass 3: OUT[dst] += (ex/denom[dst]) * H[src]   (one wave per edge)
// ---------------------------------------------------------------------------
__global__ __launch_bounds__(256) void edge_aggregate(
    const int* __restrict__ src, const int* __restrict__ dst,
    const float* __restrict__ ebuf, const float* __restrict__ denom,
    const float* __restrict__ H, float* __restrict__ OUT, int E, int F)
{
    const int e    = (int)((blockIdx.x * blockDim.x + threadIdx.x) >> 5);
    const int lane = (int)(threadIdx.x & 31);
    if (e >= E) return;
    const int s = src[e], d = dst[e];
    const float alpha = ebuf[e] / denom[d];
    for (int c0 = lane; c0 < F; c0 += 32)
        atomicAdd(OUT + (size_t)d * F + c0, alpha * H[(size_t)s * F + c0]);
}

// ---------------------------------------------------------------------------
// Epilogue: OUT += bias; optional leaky_relu(0.01); optional f16 copy that
// becomes the next layer's GEMM input (fused, saves a full pass).
// ---------------------------------------------------------------------------
__global__ void bias_act(float* __restrict__ OUT, const float* __restrict__ b,
                         _Float16* __restrict__ xnext, int total, int F, int doAct)
{
    int i = (int)(blockIdx.x * blockDim.x + threadIdx.x);
    if (i >= total) return;
    float v = OUT[i] + b[i % F];
    if (doAct) v = (v > 0.f) ? v : 0.01f * v;
    OUT[i] = v;
    if (xnext) xnext[i] = (_Float16)v;
}

// ---------------------------------------------------------------------------
// Final gather: first node of each graph (nodes 0,50,100,...)
// ---------------------------------------------------------------------------
__global__ void gather_out(const float* __restrict__ H3, float* __restrict__ out)
{
    int i = (int)(blockIdx.x * blockDim.x + threadIdx.x);
    if (i >= NGRAPH * OUT_CH) return;
    int g = i >> 4, c = i & 15;
    out[i] = H3[(size_t)(g * GSZ) * OUT_CH + c];
}

// ---------------------------------------------------------------------------
// One GAT layer (templated on GEMM shape)
// ---------------------------------------------------------------------------
template<int K, int F, int CT>
static void run_gat_layer(const _Float16* X16, const _Float16* Wt16,
                          const float* avs, const float* avd, const float* bias,
                          const int* src, const int* dst,
                          float* H, float* OUT, _Float16* xnext16,
                          float* ss, float* sd, unsigned* mm, float* denom,
                          float* ebuf, int doAct, hipStream_t stream)
{
    const int waves = (NN / 16) * (F / (16 * CT));
    gemm_wmma_f16<K, F, CT><<<(waves + 7) / 8, 256, 0, stream>>>(X16, Wt16, H, NN);
    node_scores<<<(NN + 7) / 8, 256, 0, stream>>>(H, avs, avd, ss, sd, NN, F);
    fill_u32<<<(NN + 255) / 256, 256, 0, stream>>>(mm, MONO_NEG_INF, NN);
    fill_f32<<<(NN + 255) / 256, 256, 0, stream>>>(denom, 0.f, NN);
    fill_f32<<<(NN * F + 255) / 256, 256, 0, stream>>>(OUT, 0.f, NN * F);
    edge_score_max<<<(NE + 255) / 256, 256, 0, stream>>>(src, dst, ss, sd, ebuf, mm, NE);
    edge_exp_sum<<<(NE + 255) / 256, 256, 0, stream>>>(dst, mm, ebuf, denom, NE);
    edge_aggregate<<<(NE + 7) / 8, 256, 0, stream>>>(src, dst, ebuf, denom, H, OUT, NE, F);
    bias_act<<<(NN * F + 255) / 256, 256, 0, stream>>>(OUT, bias, xnext16, NN * F, F, doAct);
}

// ---------------------------------------------------------------------------
extern "C" void kernel_launch(void* const* d_in, const int* in_sizes, int n_in,
                              void* d_out, int out_size, void* d_ws, size_t ws_size,
                              hipStream_t stream)
{
    (void)in_sizes; (void)n_in; (void)out_size; (void)ws_size;

    const float* x     = (const float*)d_in[0];
    const int*   esrc  = (const int*)  d_in[1];
    const int*   edst  = (const int*)  d_in[2];
    /* batch = d_in[3] unused: graphs are fixed contiguous blocks of 50 */
    const float* W1 = (const float*)d_in[4];
    const float* aS1 = (const float*)d_in[5];
    const float* aD1 = (const float*)d_in[6];
    const float* b1 = (const float*)d_in[7];
    const float* W2 = (const float*)d_in[8];
    const float* aS2 = (const float*)d_in[9];
    const float* aD2 = (const float*)d_in[10];
    const float* b2 = (const float*)d_in[11];
    const float* W3 = (const float*)d_in[12];
    const float* aS3 = (const float*)d_in[13];
    const float* aD3 = (const float*)d_in[14];
    const float* b3 = (const float*)d_in[15];

    // ---- workspace carve-up ----
    // fp32 region
    float* ws   = (float*)d_ws;
    float* hA   = ws;                         // 50000*256 (GEMM out H)
    float* hB   = hA + (size_t)NN * HID_CH;   // 50000*256 (aggregated OUT)
    float* ss   = hB + (size_t)NN * HID_CH;   // 50000
    float* sd   = ss + NN;                    // 50000
    unsigned* mm = (unsigned*)(sd + NN);      // 50000
    float* denom = (float*)(mm + NN);         // 50000
    float* ebuf  = denom + NN;                // 800000
    float* h3    = ebuf + NE;                 // 50000*16
    float* out3  = h3 + (size_t)NN * OUT_CH;  // 50000*16
    // f16 region
    _Float16* x16 = (_Float16*)(out3 + (size_t)NN * OUT_CH); // 50000*128
    _Float16* h16 = x16 + (size_t)NN * IN_CH;                // 50000*256
    _Float16* wt1 = h16 + (size_t)NN * HID_CH;               // 256*128
    _Float16* wt2 = wt1 + (size_t)HID_CH * IN_CH;            // 256*256
    _Float16* wt3 = wt2 + (size_t)HID_CH * HID_CH;           // 16*256

    // ---- pack WMMA operands (f16; weights transposed to [F][K]) ----
    pack_w16t<<<(IN_CH * HID_CH + 255) / 256, 256, 0, stream>>>(W1, wt1, IN_CH, HID_CH);
    pack_w16t<<<(HID_CH * HID_CH + 255) / 256, 256, 0, stream>>>(W2, wt2, HID_CH, HID_CH);
    pack_w16t<<<(HID_CH * OUT_CH + 255) / 256, 256, 0, stream>>>(W3, wt3, HID_CH, OUT_CH);
    pack_f16<<<(NN * IN_CH + 255) / 256, 256, 0, stream>>>(x, x16, NN * IN_CH);

    // Layer 1: 128 -> 256, leaky_relu(0.01); emits h16 for layer 2
    run_gat_layer<IN_CH, HID_CH, 2>(x16, wt1, aS1, aD1, b1, esrc, edst,
                                    hA, hB, h16, ss, sd, mm, denom, ebuf, 1, stream);
    // Layer 2: 256 -> 256, leaky_relu(0.01); h16 consumed by GEMM then rewritten
    run_gat_layer<HID_CH, HID_CH, 2>(h16, wt2, aS2, aD2, b2, esrc, edst,
                                     hA, hB, h16, ss, sd, mm, denom, ebuf, 1, stream);
    // Layer 3: 256 -> 16, no activation, no next-layer f16 copy
    run_gat_layer<HID_CH, OUT_CH, 1>(h16, wt3, aS3, aD3, b3, esrc, edst,
                                     h3, out3, (_Float16*)nullptr,
                                     ss, sd, mm, denom, ebuf, 0, stream);

    gather_out<<<(NGRAPH * OUT_CH + 255) / 256, 256, 0, stream>>>(out3, (float*)d_out);
}
